// Lstm_81681688035945
// MI455X (gfx1250) — compile-verified, matrix operand order checked by reading, not run on hardware
//
#include <hip/hip_runtime.h>
#include <math.h>

// Problem constants (from reference): B=64, T=32, V=10000, H=256, O=10000
#define LB 64
#define LT 32
#define LV 10000
#define LH 256
#define LO 10000
#define G4H 1024  // 4*H

typedef __attribute__((ext_vector_type(16))) _Float16 v16h;
typedef __attribute__((ext_vector_type(8)))  _Float16 v8h;
typedef __attribute__((ext_vector_type(8)))  float    v8f;

// ---------------------------------------------------------------------------
// Fragment loader for V_WMMA_F32_16X16X32_F16.
// 16-bit A-matrix 16x32 layout (ISA 7.12.2): lanes 0-15 hold M=lane with
// K=0..7 (elems 0-7) and K=16..23 (elems 8-15); lanes 16-31 hold the same M
// rows with K=8..15 / K=24..31.  B (N x K row-major, i.e. weight [out,in])
// uses the identical pattern with N in place of M.  Two 16-byte loads/lane.
// ---------------------------------------------------------------------------
__device__ __forceinline__ v16h frag_load(const _Float16* __restrict__ p,
                                          int ld, int row0, int col0) {
  const int lane = threadIdx.x & 31;
  const _Float16* base =
      p + (size_t)(row0 + (lane & 15)) * ld + col0 + ((lane >> 4) << 3);
  v16h f;
  *reinterpret_cast<v8h*>(&f)       = *reinterpret_cast<const v8h*>(base);
  *(reinterpret_cast<v8h*>(&f) + 1) = *reinterpret_cast<const v8h*>(base + 16);
  return f;
}

__device__ __forceinline__ v8f wmma_f16(v16h a, v16h b, v8f c) {
  // (neg_a, A, neg_b, B, c_mod, C, reuse_a, reuse_b)
  return __builtin_amdgcn_wmma_f32_16x16x32_f16(false, a, false, b,
                                                (short)0, c, false, false);
}

__device__ __forceinline__ float sigmoidf_fast(float x) {
  return 1.0f / (1.0f + __expf(-x));
}

// ---------------------------------------------------------------------------
// f32 -> f16 conversion (emb, W_ih, W_hh, W_out)
// ---------------------------------------------------------------------------
__global__ void f32_to_f16_kernel(const float* __restrict__ src,
                                  _Float16* __restrict__ dst, int n) {
  for (int i = blockIdx.x * blockDim.x + threadIdx.x; i < n;
       i += gridDim.x * blockDim.x)
    dst[i] = (_Float16)src[i];
}

// ---------------------------------------------------------------------------
// xg[m, g] = emb16[x[m], :] . W_ih[g, :] + b_ih[g] + b_hh[g]
// m = b*T + t in [0, 2048); g in [0, 1024).  One wave per 16x16 tile.
// Tiles: 128 (M) x 64 (N) = 8192 waves = 1024 blocks x 8 waves.
// ---------------------------------------------------------------------------
__global__ void xg_kernel(const int* __restrict__ x,
                          const _Float16* __restrict__ emb16,
                          const _Float16* __restrict__ wih16,
                          const float* __restrict__ b_ih,
                          const float* __restrict__ b_hh,
                          float* __restrict__ xg) {
  const int wave = (blockIdx.x * blockDim.x + threadIdx.x) >> 5;
  const int lane = threadIdx.x & 31;
  const int mt = wave >> 6;   // 0..127
  const int nt = wave & 63;   // 0..63
  const int m0 = mt * 16, n0 = nt * 16;

  // Gathered A row: each lane pair-group reads row m0+(lane&15) of emb[x[.]]
  const int tok = x[m0 + (lane & 15)];
  const _Float16* arow = emb16 + (size_t)tok * LH + ((lane >> 4) << 3);

  v8f acc = {};
#pragma unroll
  for (int k0 = 0; k0 < LH; k0 += 32) {
    v16h a;
    *reinterpret_cast<v8h*>(&a)       = *reinterpret_cast<const v8h*>(arow + k0);
    *(reinterpret_cast<v8h*>(&a) + 1) = *reinterpret_cast<const v8h*>(arow + k0 + 16);
    v16h b = frag_load(wih16, LH, n0, k0);
    acc = wmma_f16(a, b, acc);
  }

  const int n = n0 + (lane & 15);
  const float bias = b_ih[n] + b_hh[n];
#pragma unroll
  for (int r = 0; r < 8; ++r) {
    const int m = m0 + r + ((lane >> 4) << 3);  // C/D layout: lanes>=16 hold M=r+8
    xg[(size_t)m * G4H + n] = acc[r] + bias;
  }
}

// ---------------------------------------------------------------------------
// Persistent single-workgroup LSTM recurrence: 1024 sequential cell steps.
// 512 threads = 16 waves; wave j owns H-columns [j*16, j*16+16) for all 4
// gates and all 4 batch tiles (m=0..3).  h kept as f16 in LDS (32 KB), c kept
// entirely in registers (each lane always owns the same 32 (b,hc) cells).
// W_hh streamed from L2 each step (512 KB f16, L2-resident).
// Per step: 16 waves x (8 k-steps x (4 A-frags + 4 B-frags + 16 wmma)).
// ---------------------------------------------------------------------------
__global__ void __launch_bounds__(512)
lstm_kernel(const _Float16* __restrict__ whh16,
            const float* __restrict__ xg,
            _Float16* __restrict__ ysel,      // [T*B, H] f16, row = i*B + b
            float* __restrict__ out_hc) {     // [0:16384) h, [16384:32768) c
  __shared__ _Float16 h16[LB * LH];           // 32 KB

  const int tid  = threadIdx.x;
  const int wave = tid >> 5;                  // 0..15 == j (H-column tile)
  const int lane = tid & 31;
  const int j    = wave;
  const int hc   = j * 16 + (lane & 15);      // this lane's H column

  for (int i = tid; i < LB * LH; i += 512) h16[i] = (_Float16)0.0f;

  float c_reg[4][8];                          // c state, register-resident
#pragma unroll
  for (int m = 0; m < 4; ++m)
#pragma unroll
    for (int r = 0; r < 8; ++r) c_reg[m][r] = 0.0f;

  __syncthreads();

  for (int s = 0; s < LT * LT; ++s) {         // 1024 sequential cell steps
    const int t = s & 31;                     // inner scan position
    const int i = s >> 5;                     // outer iteration

    // ---- phase 1: gates = h @ W_hh^T (reads h16) ----
    v8f acc[4][4];                            // [m-tile][gate]
#pragma unroll
    for (int m = 0; m < 4; ++m)
#pragma unroll
      for (int g = 0; g < 4; ++g) acc[m][g] = (v8f){};

#pragma unroll
    for (int k0 = 0; k0 < LH; k0 += 32) {
      v16h a[4];
#pragma unroll
      for (int m = 0; m < 4; ++m) a[m] = frag_load(h16, LH, m * 16, k0);
#pragma unroll
      for (int g = 0; g < 4; ++g) {
        v16h b = frag_load(whh16, LH, g * 256 + j * 16, k0);
#pragma unroll
        for (int m = 0; m < 4; ++m) acc[m][g] = wmma_f16(a[m], b, acc[m][g]);
      }
    }

    __syncthreads();  // all waves done reading h16 before owners overwrite it

    // ---- phase 2: activations + state update (writes h16) ----
#pragma unroll
    for (int m = 0; m < 4; ++m) {
#pragma unroll
      for (int r = 0; r < 8; ++r) {
        const int b = m * 16 + r + ((lane >> 4) << 3);  // batch row
        const float* gx = xg + (size_t)(b * LT + t) * G4H;
        const float gi = acc[m][0][r] + gx[0 * 256 + hc];
        const float gf = acc[m][1][r] + gx[1 * 256 + hc];
        const float gg = acc[m][2][r] + gx[2 * 256 + hc];
        const float go = acc[m][3][r] + gx[3 * 256 + hc];
        const float iv = sigmoidf_fast(gi);
        const float fv = sigmoidf_fast(gf);
        const float gv = tanhf(gg);
        const float ov = sigmoidf_fast(go);
        const float cn = fv * c_reg[m][r] + iv * gv;
        const float hn = ov * tanhf(cn);
        c_reg[m][r] = cn;
        h16[b * LH + hc] = (_Float16)hn;
        if (t == i)  // ys[i] of outer iteration i
          ysel[(size_t)(i * LB + b) * LH + hc] = (_Float16)hn;
        if (s == LT * LT - 1) {  // final carry (h, c)
          out_hc[b * LH + hc]           = hn;
          out_hc[LB * LH + b * LH + hc] = cn;
        }
      }
    }

    __syncthreads();  // writes visible before next step's reads
  }
}

// ---------------------------------------------------------------------------
// y[b, t, o] = ysel[t*B+b, :] . W_out[o, :] + b_out[o]
// GEMM [2048, 256] x [10000, 256]^T.  Tiles: 128 (M) x 625 (N) = 80000 waves
// = 10000 blocks x 8 waves.  Output write is the HBM-bound part (82 MB).
// ---------------------------------------------------------------------------
__global__ void out_kernel(const _Float16* __restrict__ ysel,
                           const _Float16* __restrict__ wout16,
                           const float* __restrict__ b_out,
                           float* __restrict__ y) {
  const int wave = (blockIdx.x * blockDim.x + threadIdx.x) >> 5;
  const int lane = threadIdx.x & 31;
  const int NT = LO / 16;  // 625
  const int mt = wave / NT;
  const int nt = wave % NT;
  const int m0 = mt * 16, n0 = nt * 16;

  v8f acc = {};
#pragma unroll
  for (int k0 = 0; k0 < LH; k0 += 32) {
    v16h a = frag_load(ysel,   LH, m0, k0);
    v16h b = frag_load(wout16, LH, n0, k0);
    acc = wmma_f16(a, b, acc);
  }

  const int o = n0 + (lane & 15);
  const float bias = b_out[o];
#pragma unroll
  for (int r = 0; r < 8; ++r) {
    const int m = m0 + r + ((lane >> 4) << 3);  // ysel row = t*B + b
    const int t = m >> 6;
    const int b = m & 63;
    y[(size_t)(b * LT + t) * LO + o] = acc[r] + bias;
  }
}

// ---------------------------------------------------------------------------
// Host launcher
// ---------------------------------------------------------------------------
extern "C" void kernel_launch(void* const* d_in, const int* in_sizes, int n_in,
                              void* d_out, int out_size, void* d_ws,
                              size_t ws_size, hipStream_t stream) {
  (void)in_sizes; (void)n_in; (void)out_size; (void)ws_size;

  const int*   x     = (const int*)  d_in[0];
  const float* emb   = (const float*)d_in[1];
  const float* W_ih  = (const float*)d_in[2];
  const float* W_hh  = (const float*)d_in[3];
  const float* b_ih  = (const float*)d_in[4];
  const float* b_hh  = (const float*)d_in[5];
  const float* W_out = (const float*)d_in[6];
  const float* b_out = (const float*)d_in[7];
  float* out = (float*)d_out;  // [h(16384) | c(16384) | y(20480000)] f32

  // Workspace layout (bytes, all 16B-aligned)
  char* ws = (char*)d_ws;
  _Float16* emb16  = (_Float16*)(ws + 0);         //  5,120,000
  _Float16* wih16  = (_Float16*)(ws + 5120000);   //    524,288
  _Float16* whh16  = (_Float16*)(ws + 5644288);   //    524,288
  _Float16* wout16 = (_Float16*)(ws + 6168576);   //  5,120,000
  float*    xg     = (float*)   (ws + 11288576);  //  8,388,608
  _Float16* ysel   = (_Float16*)(ws + 19677184);  //  1,048,576  -> ~20.7 MB

  // 1) precision conversion
  f32_to_f16_kernel<<<256, 256, 0, stream>>>(emb,   emb16,  LV * LH);
  f32_to_f16_kernel<<<64,  256, 0, stream>>>(W_ih,  wih16,  G4H * LH);
  f32_to_f16_kernel<<<64,  256, 0, stream>>>(W_hh,  whh16,  G4H * LH);
  f32_to_f16_kernel<<<256, 256, 0, stream>>>(W_out, wout16, LO * LH);

  // 2) input gates: gather + GEMM  (8192 tiles / 8 waves per block)
  xg_kernel<<<1024, 256, 0, stream>>>(x, emb16, wih16, b_ih, b_hh, xg);

  // 3) sequential recurrence: one persistent workgroup on one WGP
  lstm_kernel<<<1, 512, 0, stream>>>(whh16, xg, ysel, out);

  // 4) output projection (80000 tiles / 8 waves per block)
  out_kernel<<<10000, 256, 0, stream>>>(ysel, wout16, b_out,
                                        out + 2 * LB * LH);
}
